// Cross_Attention_13580686590370
// MI455X (gfx1250) — compile-verified, matrix operand order checked by reading, not run on hardware
//
#include <hip/hip_runtime.h>
#include <math.h>

// ---------------------------------------------------------------------------
// Cross channel-attention block for MI455X (gfx1250, wave32, WMMA).
// The 1x1 and dense 3x3 convs run as bf16 WMMA GEMMs (v_wmma_f32_16x16x32_bf16), fp32 accum.
// GEMM: block 256 thr = 8 waves; block tile 256x64; wave tile 64x32 (4x2 WMMA).
// All strides/K are compile-time -> fully unrolled K loop, global_load with
// immediate offsets. kv 3x3 conv uses a zero-padded activation buffer so the
// 9-tap implicit-im2col GEMM has NO border predication (no EXEC toggles).
// ---------------------------------------------------------------------------

typedef __bf16 bf16;
typedef __attribute__((ext_vector_type(16))) __bf16 v16bf;
typedef __attribute__((ext_vector_type(8)))  __bf16 v8bf;
typedef __attribute__((ext_vector_type(8)))  float  v8f;

static constexpr int DIM   = 192;
static constexpr int HEADS = 8;
static constexpr int BATCH = 4;
static constexpr int HH    = 128;
static constexpr int WW    = 128;
static constexpr int HW    = HH * WW;        // 16384
static constexpr int NPIX  = BATCH * HW;     // 65536
static constexpr int CP    = DIM / HEADS;    // 24
static constexpr int C3    = 3 * DIM;        // 576
static constexpr int C2    = 2 * DIM;        // 384
static constexpr int HP    = HH + 2;         // 130 (padded)
static constexpr int WP    = WW + 2;         // 130 (padded)

#define DEVFN __device__ __forceinline__

union Frag { v16bf v; v8bf h[2]; };

// A/B fragment (16x32 bf16): lane holds row (lane&15); K chunks {kh..kh+7,
// kh+16..kh+23}, kh = 8*(lane>>4), per CDNA5 ISA 7.12.2. p must already point
// at (row, kh) of the tile: two contiguous 16B loads.
DEVFN v16bf ld_frag(const bf16* __restrict__ p) {
  Frag f;
  f.h[0] = *(const v8bf*)(p);
  f.h[1] = *(const v8bf*)(p + 16);
  return f.v;
}

DEVFN v8f wmma_bf(v16bf a, v16bf b, v8f c) {
  return __builtin_amdgcn_wmma_f32_16x16x32_bf16(
      /*neg_a=*/false, a, /*neg_b=*/false, b,
      /*c_mod=*/(short)0, c, /*reuse_a=*/false, /*reuse_b=*/false);
}

// ---------------------------------------------------------------------------
// Layout conversion kernels
// ---------------------------------------------------------------------------

// NCHW fp32 -> [B*H*W, C] bf16 (pixel-major activations)
__global__ void cvt_nchw(const float* __restrict__ src, bf16* __restrict__ dst, int C) {
  size_t idx = (size_t)blockIdx.x * blockDim.x + threadIdx.x;
  size_t total = (size_t)NPIX * C;
  if (idx >= total) return;
  size_t chw = (size_t)C * HW;
  int b = (int)(idx / chw);
  size_t r = idx - (size_t)b * chw;
  int c = (int)(r / HW);
  int pix = (int)(r % HW);
  dst[((size_t)b * HW + pix) * C + c] = (bf16)src[idx];
}

// NCHW fp32 -> zero-padded [B][H+2][W+2][C] bf16 (for the 3x3 kv conv)
__global__ void cvt_nchw_pad(const float* __restrict__ src, bf16* __restrict__ dst) {
  size_t idx = (size_t)blockIdx.x * blockDim.x + threadIdx.x;
  size_t total = (size_t)BATCH * HP * WP * DIM;
  if (idx >= total) return;
  int c = (int)(idx % DIM);
  size_t r = idx / DIM;
  int xp = (int)(r % WP); r /= WP;
  int yp = (int)(r % HP);
  int b  = (int)(r / HP);
  int y = yp - 1, x = xp - 1;
  bf16 v = (bf16)0.0f;
  if ((unsigned)y < (unsigned)HH && (unsigned)x < (unsigned)WW)
    v = (bf16)src[(((size_t)b * DIM + c) * HH + y) * WW + x];
  dst[idx] = v;
}

__global__ void cvt_f32_bf16(const float* __restrict__ s, bf16* __restrict__ d, int n) {
  int i = blockIdx.x * blockDim.x + threadIdx.x;
  if (i < n) d[i] = (bf16)s[i];
}

// kv_w OIHW (384,192,3,3) -> tap-major bf16 [9][384][192]
__global__ void cvt_kvw(const float* __restrict__ s, bf16* __restrict__ d) {
  int i = blockIdx.x * blockDim.x + threadIdx.x;
  if (i >= 9 * C2 * DIM) return;
  int tap = i / (C2 * DIM);
  int r   = i % (C2 * DIM);            // o*192 + ic
  d[i] = (bf16)s[(size_t)r * 9 + tap];
}

// ---------------------------------------------------------------------------
// Templated WMMA GEMM:  C[M,N](bf16) = A[M,K](bf16) * Bt[N,K]^T + bias[N]
// Two compile-time phases for the channel-concat (no pointer selects -> no
// FLAT fallback). N comes from gridDim.x*64.
// ---------------------------------------------------------------------------
template <int LDA0, int ACOL0, int KSPLIT, int LDA1, int ACOL1, int K, int LDC>
__global__ __launch_bounds__(256) void gemm_bf16_t(
    const bf16* __restrict__ A0, const bf16* __restrict__ A1,
    const bf16* __restrict__ Bt, const float* __restrict__ bias,
    bf16* __restrict__ C) {
  int lane = threadIdx.x & 31, wave = threadIdx.x >> 5;
  int Mbase = blockIdx.y * 256 + (wave & 3) * 64;
  int Nbase = blockIdx.x * 64 + (wave >> 2) * 32;
  int rl = lane & 15;
  int kh = (lane >> 4) << 3;

  v8f acc[4][2] = {};

  const bf16* pb = Bt + (size_t)(Nbase + rl) * K + kh;

  {  // phase 0: k in [0, KSPLIT) from A0
    const bf16* pa = A0 + (size_t)(Mbase + rl) * LDA0 + ACOL0 + kh;
#pragma unroll
    for (int k0 = 0; k0 < KSPLIT; k0 += 32) {
      v16bf b0 = ld_frag(pb + k0);
      v16bf b1 = ld_frag(pb + k0 + 16 * K);
      __builtin_prefetch(pa + k0 + 256);     // global_prefetch_b8
#pragma unroll
      for (int mi = 0; mi < 4; ++mi) {
        v16bf a = ld_frag(pa + k0 + mi * (16 * LDA0));
        acc[mi][0] = wmma_bf(a, b0, acc[mi][0]);
        acc[mi][1] = wmma_bf(a, b1, acc[mi][1]);
      }
    }
  }
  if constexpr (KSPLIT < K) {  // phase 1: k in [KSPLIT, K) from A1
    const bf16* pa = A1 + (size_t)(Mbase + rl) * LDA1 + ACOL1 + kh;
#pragma unroll
    for (int k0 = KSPLIT; k0 < K; k0 += 32) {
      v16bf b0 = ld_frag(pb + k0);
      v16bf b1 = ld_frag(pb + k0 + 16 * K);
      __builtin_prefetch(pa + (k0 - KSPLIT) + 256);
#pragma unroll
      for (int mi = 0; mi < 4; ++mi) {
        v16bf a = ld_frag(pa + (k0 - KSPLIT) + mi * (16 * LDA1));
        acc[mi][0] = wmma_bf(a, b0, acc[mi][0]);
        acc[mi][1] = wmma_bf(a, b1, acc[mi][1]);
      }
    }
  }

  int cn = Nbase + rl;
  int rb = Mbase + kh;   // kh == 8*(lane>>4): C/D row offset per ISA layout
#pragma unroll
  for (int mi = 0; mi < 4; ++mi)
#pragma unroll
    for (int ni = 0; ni < 2; ++ni) {
      int coln = cn + ni * 16;
      float bv = bias[coln];
#pragma unroll
      for (int j = 0; j < 8; ++j) {
        int row = rb + mi * 16 + j;
        C[(size_t)row * LDC + coln] = (bf16)(acc[mi][ni][j] + bv);
      }
    }
}

// ---------------------------------------------------------------------------
// kv 3x3 dense conv as implicit-im2col over the zero-padded activation:
// 9 shifted GEMMs (K=192) with plain unpredicated loads; each tap is a
// constant byte offset from 4 per-strip base pointers.
// ---------------------------------------------------------------------------
__global__ __launch_bounds__(256) void gemm_kvconv(
    const bf16* __restrict__ Xp /*[B][130][130][192]*/,
    const bf16* __restrict__ Wt /*[9][384][192]*/,
    const float* __restrict__ bias, bf16* __restrict__ C /*[NPIX,384]*/) {
  int lane = threadIdx.x & 31, wave = threadIdx.x >> 5;
  int Mbase = blockIdx.y * 256 + (wave & 3) * 64;
  int Nbase = blockIdx.x * 64 + (wave >> 2) * 32;
  int rl = lane & 15;
  int kh = (lane >> 4) << 3;
  int rbase = Mbase + rl;

  // Per-strip base pointer at the (unshifted) pixel inside the padded image.
  const bf16* pa[4];
#pragma unroll
  for (int mi = 0; mi < 4; ++mi) {
    int rr = rbase + mi * 16;
    int bb = rr >> 14;
    int rem = rr & (HW - 1);
    int y = rem >> 7, x = rem & 127;
    pa[mi] = Xp + ((size_t)(bb * HP + y + 1) * WP + (x + 1)) * DIM + kh;
  }

  v8f acc[4][2] = {};

  for (int tap = 0; tap < 9; ++tap) {
    int doff = ((tap / 3 - 1) * WP + (tap % 3 - 1)) * DIM;   // tap shift (elems)
    const bf16* pbt = Wt + (size_t)tap * C2 * DIM + (size_t)(Nbase + rl) * DIM + kh;
#pragma unroll
    for (int k0 = 0; k0 < DIM; k0 += 32) {
      v16bf b0 = ld_frag(pbt + k0);
      v16bf b1 = ld_frag(pbt + k0 + 16 * DIM);
#pragma unroll
      for (int mi = 0; mi < 4; ++mi) {
        v16bf a = ld_frag(pa[mi] + doff + k0);
        acc[mi][0] = wmma_bf(a, b0, acc[mi][0]);
        acc[mi][1] = wmma_bf(a, b1, acc[mi][1]);
      }
    }
  }

  int cn = Nbase + rl;
  int rb = Mbase + kh;
#pragma unroll
  for (int mi = 0; mi < 4; ++mi)
#pragma unroll
    for (int ni = 0; ni < 2; ++ni) {
      int coln = cn + ni * 16;
      float bv = bias[coln];
#pragma unroll
      for (int j = 0; j < 8; ++j) {
        int row = rb + mi * 16 + j;
        C[(size_t)row * C2 + coln] = (bf16)(acc[mi][ni][j] + bv);
      }
    }
}

// ---------------------------------------------------------------------------
// proj 1x1 GEMM with fp32 NCHW scatter store (final output)
// ---------------------------------------------------------------------------
__global__ __launch_bounds__(256) void gemm_proj(
    const bf16* __restrict__ A, const bf16* __restrict__ Bt,
    const float* __restrict__ bias, float* __restrict__ Out) {
  int lane = threadIdx.x & 31, wave = threadIdx.x >> 5;
  int Mbase = blockIdx.y * 256 + (wave & 3) * 64;
  int Nbase = blockIdx.x * 64 + (wave >> 2) * 32;
  int rl = lane & 15;
  int kh = (lane >> 4) << 3;

  v8f acc[4][2] = {};

  const bf16* pb = Bt + (size_t)(Nbase + rl) * DIM + kh;
  const bf16* pa = A + (size_t)(Mbase + rl) * DIM + kh;
#pragma unroll
  for (int k0 = 0; k0 < DIM; k0 += 32) {
    v16bf b0 = ld_frag(pb + k0);
    v16bf b1 = ld_frag(pb + k0 + 16 * DIM);
#pragma unroll
    for (int mi = 0; mi < 4; ++mi) {
      v16bf a = ld_frag(pa + k0 + mi * (16 * DIM));
      acc[mi][0] = wmma_bf(a, b0, acc[mi][0]);
      acc[mi][1] = wmma_bf(a, b1, acc[mi][1]);
    }
  }

  int cn = Nbase + rl;
  int rb = Mbase + kh;
#pragma unroll
  for (int mi = 0; mi < 4; ++mi)
#pragma unroll
    for (int ni = 0; ni < 2; ++ni) {
      int coln = cn + ni * 16;
      float bv = bias[coln];
#pragma unroll
      for (int j = 0; j < 8; ++j) {
        int row = rb + mi * 16 + j;
        int b = row >> 14;
        int sp = row & (HW - 1);
        Out[((size_t)b * DIM + coln) * HW + sp] = acc[mi][ni][j] + bv;
      }
    }
}

// ---------------------------------------------------------------------------
// Grouped 3x3 conv (groups=192, 3 in / 3 out per group) — VALU (2 GF)
// ---------------------------------------------------------------------------
__global__ void qdw_conv(const bf16* __restrict__ in, const float* __restrict__ w,
                         const float* __restrict__ bias, bf16* __restrict__ outb) {
  size_t idx = (size_t)blockIdx.x * blockDim.x + threadIdx.x;
  if (idx >= (size_t)NPIX * C3) return;
  int o = (int)(idx % C3);
  size_t p = idx / C3;
  int g3 = (o / 3) * 3;
  int rem = (int)(p & (HW - 1));
  int y = rem >> 7, x = rem & 127;
  float s = bias[o];
#pragma unroll
  for (int ky = 0; ky < 3; ++ky) {
    int yy = y + ky - 1;
    if ((unsigned)yy >= (unsigned)HH) continue;
#pragma unroll
    for (int kx = 0; kx < 3; ++kx) {
      int xx = x + kx - 1;
      if ((unsigned)xx >= (unsigned)WW) continue;
      const bf16* ip = in + ((ptrdiff_t)p + (ky - 1) * WW + (kx - 1)) * C3 + g3;
      const float* wp = w + (size_t)o * 27 + ky * 3 + kx;
      s += wp[0]  * (float)ip[0];
      s += wp[9]  * (float)ip[1];
      s += wp[18] * (float)ip[2];
    }
  }
  outb[idx] = (bf16)s;
}

// Depthwise 3x3 conv (384 channels) — VALU (0.45 GF)
__global__ void kvdw_conv(const bf16* __restrict__ in, const float* __restrict__ w,
                          const float* __restrict__ bias, bf16* __restrict__ outb) {
  size_t idx = (size_t)blockIdx.x * blockDim.x + threadIdx.x;
  if (idx >= (size_t)NPIX * C2) return;
  int o = (int)(idx % C2);
  size_t p = idx / C2;
  int rem = (int)(p & (HW - 1));
  int y = rem >> 7, x = rem & 127;
  float s = bias[o];
#pragma unroll
  for (int ky = 0; ky < 3; ++ky) {
    int yy = y + ky - 1;
    if ((unsigned)yy >= (unsigned)HH) continue;
#pragma unroll
    for (int kx = 0; kx < 3; ++kx) {
      int xx = x + kx - 1;
      if ((unsigned)xx >= (unsigned)WW) continue;
      s += w[(size_t)o * 9 + ky * 3 + kx] *
           (float)in[((ptrdiff_t)p + (ky - 1) * WW + (kx - 1)) * C2 + o];
    }
  }
  outb[idx] = (bf16)s;
}

// ---------------------------------------------------------------------------
// Inverse L2 norms per (tensor, b, channel): inv = 1/max(||row||, 1e-12)
// ---------------------------------------------------------------------------
__global__ __launch_bounds__(256) void inv_norm(
    const bf16* __restrict__ qsrc,  // qkv1, q part = cols 0..191 of 576
    const bf16* __restrict__ ksrc,  // knew [pix,192]
    const bf16* __restrict__ vsrc,  // vnew [pix,192]
    float* __restrict__ invn) {
  __shared__ float red[256];
  int t = blockIdx.y;  // 0=q 1=k 2=v
  int b = blockIdx.x / DIM, c = blockIdx.x % DIM;
  const bf16* src;
  int ld;
  if (t == 0) { src = qsrc; ld = C3; }
  else if (t == 1) { src = ksrc; ld = DIM; }
  else { src = vsrc; ld = DIM; }
  const bf16* p = src + (size_t)b * HW * ld + c;
  float s = 0.f;
  for (int i = threadIdx.x; i < HW; i += 256) {
    float v = (float)p[(size_t)i * ld];
    s += v * v;
  }
  red[threadIdx.x] = s;
  __syncthreads();
  for (int w = 128; w > 0; w >>= 1) {
    if ((int)threadIdx.x < w) red[threadIdx.x] += red[threadIdx.x + w];
    __syncthreads();
  }
  if (threadIdx.x == 0)
    invn[(size_t)t * BATCH * DIM + b * DIM + c] = 1.0f / fmaxf(sqrtf(red[0]), 1e-12f);
}

// ---------------------------------------------------------------------------
// Raw channel-attention scores S[b,h,c,d] = sum_n q[c,n]*k[d,n]
// ---------------------------------------------------------------------------
__global__ __launch_bounds__(256) void attn_qk(
    const bf16* __restrict__ qkv1, const bf16* __restrict__ knew,
    float* __restrict__ araw) {
  __shared__ float qs[CP * 65];
  __shared__ float ks[CP * 65];
  int bh = blockIdx.x, b = bh >> 3, h = bh & 7;
  int co = h * CP;
  int tid = threadIdx.x;
  float accp[3] = {0.f, 0.f, 0.f};
  for (int t0 = 0; t0 < HW; t0 += 64) {
    for (int i = tid; i < CP * 64; i += 256) {
      int c = i >> 6, j = i & 63;
      size_t pix = (size_t)b * HW + t0 + j;
      qs[c * 65 + j] = (float)qkv1[pix * C3 + co + c];
      ks[c * 65 + j] = (float)knew[pix * DIM + co + c];
    }
    __syncthreads();
#pragma unroll
    for (int m = 0; m < 3; ++m) {
      int pr = tid + m * 256;
      if (pr < CP * CP) {
        int c = pr / CP, d = pr % CP;
        float s = 0.f;
#pragma unroll
        for (int j = 0; j < 64; ++j) s += qs[c * 65 + j] * ks[d * 65 + j];
        accp[m] += s;
      }
    }
    __syncthreads();
  }
  for (int m = 0; m < 3; ++m) {
    int pr = tid + m * 256;
    if (pr < CP * CP) araw[(size_t)bh * CP * CP + pr] = accp[m];
  }
}

// Scale by temperature & inverse norms, softmax over d, fold in inv_v.
__global__ void attn_softmax(const float* __restrict__ araw,
                             const float* __restrict__ temp,
                             const float* __restrict__ invn,
                             float* __restrict__ attnS) {
  int r = blockIdx.x * blockDim.x + threadIdx.x;
  if (r >= BATCH * DIM) return;  // 768 rows (b, h, c)
  int b = r / DIM, hc = r % DIM, h = hc / CP, c = hc % CP;
  const float* iq = invn;
  const float* ik = invn + BATCH * DIM;
  const float* iv = invn + 2 * BATCH * DIM;
  float tm = temp[h];
  float qi = iq[b * DIM + h * CP + c];
  const float* rp = araw + ((size_t)(b * HEADS + h) * CP + c) * CP;
  float row[CP];
  float mx = -3.4e38f;
  for (int d = 0; d < CP; ++d) {
    float s = rp[d] * tm * qi * ik[b * DIM + h * CP + d];
    row[d] = s;
    mx = fmaxf(mx, s);
  }
  float sum = 0.f;
  for (int d = 0; d < CP; ++d) {
    row[d] = __expf(row[d] - mx);
    sum += row[d];
  }
  float rs = 1.0f / sum;
  float* op = attnS + ((size_t)(b * HEADS + h) * CP + c) * CP;
  for (int d = 0; d < CP; ++d)
    op[d] = row[d] * rs * iv[b * DIM + h * CP + d];
}

// out[p, h*24+c] = sum_d attnS[b,h,c,d] * v[p, h*24+d]
__global__ __launch_bounds__(256) void attn_apply(
    const float* __restrict__ attnS, const bf16* __restrict__ vnew,
    bf16* __restrict__ aout) {
  __shared__ float As[HEADS * CP * CP];  // 4608 floats = 18 KB
  int p0 = blockIdx.x * 32;
  int b = p0 >> 14;
  for (int i = threadIdx.x; i < HEADS * CP * CP; i += 256)
    As[i] = attnS[(size_t)b * HEADS * CP * CP + i];
  __syncthreads();
  int head = threadIdx.x & 7, ps = threadIdx.x >> 3;
  int p = p0 + ps;
  float vv[CP];
  const bf16* vp = vnew + (size_t)p * DIM + head * CP;
#pragma unroll
  for (int d = 0; d < CP; ++d) vv[d] = (float)vp[d];
  const float* Ap = As + head * CP * CP;
  bf16* op = aout + (size_t)p * DIM + head * CP;
#pragma unroll
  for (int c = 0; c < CP; ++c) {
    float s = 0.f;
#pragma unroll
    for (int d = 0; d < CP; ++d) s += Ap[c * CP + d] * vv[d];
    op[c] = (bf16)s;
  }
}

// ---------------------------------------------------------------------------
// Host launcher
// ---------------------------------------------------------------------------
extern "C" void kernel_launch(void* const* d_in, const int* in_sizes, int n_in,
                              void* d_out, int out_size, void* d_ws, size_t ws_size,
                              hipStream_t stream) {
  (void)in_sizes; (void)n_in; (void)out_size; (void)ws_size;
  const float* x      = (const float*)d_in[0];
  const float* xmask  = (const float*)d_in[1];
  const float* temp   = (const float*)d_in[2];
  const float* q_w    = (const float*)d_in[3];
  const float* q_b    = (const float*)d_in[4];
  const float* qdw_w  = (const float*)d_in[5];
  const float* qdw_b  = (const float*)d_in[6];
  const float* kv_w   = (const float*)d_in[7];
  const float* kv_b   = (const float*)d_in[8];
  const float* kvdw_w = (const float*)d_in[9];
  const float* kvdw_b = (const float*)d_in[10];
  const float* newk_w = (const float*)d_in[11];
  const float* newk_b = (const float*)d_in[12];
  const float* newv_w = (const float*)d_in[13];
  const float* newv_b = (const float*)d_in[14];
  const float* proj_w = (const float*)d_in[15];
  const float* proj_b = (const float*)d_in[16];
  float* out = (float*)d_out;

  char* ws = (char*)d_ws;
  size_t off = 0;
  auto alloc = [&](size_t bytes) -> void* {
    void* p = (void*)(ws + off);
    off += (bytes + 255) & ~(size_t)255;
    return p;
  };

  bf16* xbf   = (bf16*)alloc((size_t)NPIX * DIM * 2);
  bf16* xmp   = (bf16*)alloc((size_t)BATCH * HP * WP * DIM * 2);  // padded x_mask
  bf16* qkv0  = (bf16*)alloc((size_t)NPIX * C3 * 2);
  bf16* qkv1  = (bf16*)alloc((size_t)NPIX * C3 * 2);
  bf16* kv0   = (bf16*)alloc((size_t)NPIX * C2 * 2);
  bf16* kv1   = (bf16*)alloc((size_t)NPIX * C2 * 2);
  bf16* knew  = (bf16*)alloc((size_t)NPIX * DIM * 2);
  bf16* vnew  = (bf16*)alloc((size_t)NPIX * DIM * 2);
  bf16* aout  = (bf16*)alloc((size_t)NPIX * DIM * 2);
  bf16* wq    = (bf16*)alloc((size_t)C3 * DIM * 2);
  bf16* wkv   = (bf16*)alloc((size_t)9 * C2 * DIM * 2);
  bf16* wnk   = (bf16*)alloc((size_t)DIM * C2 * 2);
  bf16* wnv   = (bf16*)alloc((size_t)DIM * C2 * 2);
  bf16* wpj   = (bf16*)alloc((size_t)DIM * DIM * 2);
  float* invn = (float*)alloc((size_t)3 * BATCH * DIM * 4);
  float* araw = (float*)alloc((size_t)BATCH * HEADS * CP * CP * 4);
  float* attnS= (float*)alloc((size_t)BATCH * HEADS * CP * CP * 4);

  // Layout / precision conversions
  {
    size_t tot = (size_t)NPIX * DIM;
    cvt_nchw<<<(int)((tot + 255) / 256), 256, 0, stream>>>(x, xbf, DIM);
    size_t totp = (size_t)BATCH * HP * WP * DIM;
    cvt_nchw_pad<<<(int)((totp + 255) / 256), 256, 0, stream>>>(xmask, xmp);
  }
  cvt_f32_bf16<<<(C3 * DIM + 255) / 256, 256, 0, stream>>>(q_w, wq, C3 * DIM);
  cvt_kvw<<<(9 * C2 * DIM + 255) / 256, 256, 0, stream>>>(kv_w, wkv);
  cvt_f32_bf16<<<(DIM * C2 + 255) / 256, 256, 0, stream>>>(newk_w, wnk, DIM * C2);
  cvt_f32_bf16<<<(DIM * C2 + 255) / 256, 256, 0, stream>>>(newv_w, wnv, DIM * C2);
  cvt_f32_bf16<<<(DIM * DIM + 255) / 256, 256, 0, stream>>>(proj_w, wpj, DIM * DIM);

  // q branch: 1x1 conv (WMMA GEMM) then grouped 3x3 conv
  gemm_bf16_t<DIM, 0, DIM, DIM, 0, DIM, C3>
      <<<dim3(C3 / 64, NPIX / 256), 256, 0, stream>>>(xbf, xbf, wq, q_b, qkv0);
  qdw_conv<<<(int)((size_t)NPIX * C3 / 256), 256, 0, stream>>>(qkv0, qdw_w, qdw_b, qkv1);

  // kv branch: dense 3x3 conv (padded implicit im2col WMMA) then depthwise 3x3
  gemm_kvconv<<<dim3(C2 / 64, NPIX / 256), 256, 0, stream>>>(xmp, wkv, kv_b, kv0);
  kvdw_conv<<<(int)((size_t)NPIX * C2 / 256), 256, 0, stream>>>(kv0, kvdw_w, kvdw_b, kv1);

  // newk = 1x1 conv over concat(k, k_mask); newv over concat(v, v_mask)
  gemm_bf16_t<C3, DIM, DIM, C2, 0, C2, DIM>
      <<<dim3(DIM / 64, NPIX / 256), 256, 0, stream>>>(qkv1, kv1, wnk, newk_b, knew);
  gemm_bf16_t<C3, 2 * DIM, DIM, C2, DIM, C2, DIM>
      <<<dim3(DIM / 64, NPIX / 256), 256, 0, stream>>>(qkv1, kv1, wnv, newv_b, vnew);

  // l2 norms (folded), channel attention, softmax, apply
  inv_norm<<<dim3(BATCH * DIM, 3), 256, 0, stream>>>(qkv1, knew, vnew, invn);
  attn_qk<<<BATCH * HEADS, 256, 0, stream>>>(qkv1, knew, araw);
  attn_softmax<<<3, 256, 0, stream>>>(araw, temp, invn, attnS);
  attn_apply<<<NPIX / 32, 256, 0, stream>>>(attnS, vnew, aout);

  // proj 1x1 conv -> fp32 NCHW output
  gemm_proj<<<dim3(DIM / 64, NPIX / 256), 256, 0, stream>>>(aout, wpj, proj_b, out);
}
